// N3Aggregation2D_55018531062326
// MI455X (gfx1250) — matrix operand order repaired and checked by reading, not compiled, not run on hardware
//
#include <hip/hip_runtime.h>

// ---------------------------------------------------------------------------
// N3Aggregation2D for MI455X (gfx1250, wave32).
//  GEMM1 (distance, precision-critical)  : v_wmma_f32_16x16x4_f32
//  GEMM2 (weighted gather, 87% of FLOPs) : v_wmma_f32_16x16x32_f16
//  GEMM2 B-operand staged into LDS by the Tensor Data Mover (tensor_load_to_lds
//  + s_wait_tensorcnt), one 896x32 f16 tile per K-chunk.
//  All tile counts are padded so every wave runs exactly 7 WMMA tiles ->
//  no EXEC masking anywhere near WMMA (ISA: EXEC must be all ones).
// ---------------------------------------------------------------------------

typedef __attribute__((ext_vector_type(2)))  float        v2f;
typedef __attribute__((ext_vector_type(8)))  float        v8f;
typedef __attribute__((ext_vector_type(16))) _Float16     v16h;
typedef __attribute__((ext_vector_type(4)))  unsigned int u32x4;
typedef __attribute__((ext_vector_type(4)))  int          i32x4;
typedef __attribute__((ext_vector_type(8)))  int          i32x8;

#define T_    2
#define C_    8
#define E_    8
#define H_    64
#define W_    64
#define HP_   66
#define WP_   66
#define KK_   7
#define NW_   57          // nh == nw == 57
#define NQ_   3249        // 57*57
#define NO_   841         // 29*29
#define NOP1  896         // 56*16 : GEMM1 N padding (uniform 7 tiles/wave)
#define NOP2  864         // 27*32 : GEMM2 K padding (o dimension)
#define DK_   800         // PS*PS*E == C*PS*PS
#define ND2   896         // 56*16 : GEMM2 N padding (d dimension)
#define NQT   204         // ceil(3249/16)

__device__ __forceinline__ float wred_sum(float v) {
  #pragma unroll
  for (int m = 16; m; m >>= 1) v += __shfl_xor(v, m, 32);
  return v;
}
__device__ __forceinline__ float wred_max(float v) {
  #pragma unroll
  for (int m = 16; m; m >>= 1) v = fmaxf(v, __shfl_xor(v, m, 32));
  return v;
}
__device__ __forceinline__ void atomAddF(float* p, float v) {
  __hip_atomic_fetch_add(p, v, __ATOMIC_RELAXED, __HIP_MEMORY_SCOPE_AGENT);
}

// ---------------------------------------------------------------------------
// Prep: materialize context-patch matrices once (L2-resident afterwards).
//   xc   [T][896][800] f32 : d = p*8+e   (B of GEMM1, padded rows zero)
//   xc2  [T][896]      f32 : row squared norms
//   xcpT [T][896][864] f16 : [d][o] TRANSPOSED (B of GEMM2, TDM source tile)
// ---------------------------------------------------------------------------
__global__ __launch_bounds__(256) void n3_prep_kernel(
    const float* __restrict__ x, const float* __restrict__ xe,
    float* __restrict__ xc, float* __restrict__ xc2, _Float16* __restrict__ xcpT)
{
  __shared__ float red[8];
  const int o = blockIdx.x;            // 0..895
  const int t = blockIdx.y;
  const int tid = threadIdx.x;
  const int oh = o / 29, ow = o % 29;  // patch top-left in padded frame

  // ---- xc row + squared norm ----
  float* xcrow = xc + ((size_t)t * NOP1 + o) * DK_;
  float part = 0.f;
  if (o < NO_) {
    for (int i = tid; i < DK_; i += 256) {
      int p = i >> 3, e = i & 7;
      int r = oh + p / 10, c = ow + p % 10;
      float v = 0.f;
      if (r >= 1 && r <= H_ && c >= 1 && c <= W_)
        v = xe[(((size_t)t * E_ + e) * H_ + (r - 1)) * W_ + (c - 1)];
      xcrow[i] = v;
      part += v * v;
    }
  } else {
    for (int i = tid; i < DK_; i += 256) xcrow[i] = 0.f;
  }
  // ---- xcpT column o (transposed so TDM tiles are contiguous rows) ----
  if (o < NOP2) {
    _Float16* col = xcpT + (size_t)t * ND2 * NOP2 + o;
    for (int d = tid; d < ND2; d += 256) {
      float v = 0.f;
      if (o < NO_ && d < DK_) {
        int cch = d / 100, p = d % 100;
        int r = oh + p / 10, c = ow + p % 10;
        if (r >= 1 && r <= H_ && c >= 1 && c <= W_)
          v = x[(((size_t)t * C_ + cch) * H_ + (r - 1)) * W_ + (c - 1)];
      }
      col[(size_t)d * NOP2] = (_Float16)v;
    }
  }
  float s = wred_sum(part);
  if ((tid & 31) == 0) red[tid >> 5] = s;
  __syncthreads();
  if (tid == 0) {
    float tot = 0.f;
    #pragma unroll
    for (int i = 0; i < 8; ++i) tot += red[i];
    xc2[t * NOP1 + o] = tot;
  }
}

// ---------------------------------------------------------------------------
// Fused: 16 query rows per block (256 threads = 8 waves, wave32).
// ---------------------------------------------------------------------------
__global__ __launch_bounds__(256) void n3_fused_kernel(
    const float* __restrict__ ye, const float* __restrict__ lt,
    const float* __restrict__ xc, const float* __restrict__ xc2,
    const _Float16* __restrict__ xcpT, float* __restrict__ vid)
{
  // union: As (16x800 f32, GEMM1 A) / Bs (896x32 f16, GEMM2 B chunk via TDM)
  __shared__ __align__(128) unsigned char AsBs[ND2 * 32 * 2];   // 57344 B
  __shared__ __align__(16) float    Lg[16 * NOP1];              // 57344 B logits
  __shared__ __align__(16) _Float16 Wk[16 * NOP2];              // 27648 B W_k
  __shared__ float yq2_s[16];
  __shared__ float temp_s[16];

  float*    As = (float*)AsBs;
  _Float16* Bs = (_Float16*)AsBs;

  const int t     = blockIdx.y;
  const int qbase = blockIdx.x * 16;
  const int tid   = threadIdx.x;
  const int wave  = tid >> 5, lane = tid & 31;
  const int lhi   = lane >> 4, llo = lane & 15;

  // ---- phase 1: stage query tile, temperature, norms ----
  for (int i = tid; i < 16 * DK_; i += 256) {
    int rr = i / DK_, d = i % DK_;
    int q = qbase + rr;
    float v = 0.f;
    if (q < NQ_) {
      int p = d >> 3, e = d & 7;
      int qh = q / NW_, qw = q % NW_;
      int r = qh + p / 10, c = qw + p % 10;
      if (r >= 1 && r <= H_ && c >= 1 && c <= W_)
        v = ye[(((size_t)t * E_ + e) * H_ + (r - 1)) * W_ + (c - 1)];
    }
    As[i] = v;
  }
  if (tid < 16) {                        // per-row temperature = exp(mean(lt patch))
    int q = qbase + tid;
    float s = 0.f;
    if (q < NQ_) {
      int qh = q / NW_, qw = q % NW_;
      for (int p = 0; p < 100; ++p) {
        int r = qh + p / 10, c = qw + p % 10;
        if (r >= 1 && r <= H_ && c >= 1 && c <= W_)
          s += lt[((size_t)t * H_ + (r - 1)) * W_ + (c - 1)];
      }
    }
    temp_s[tid] = __expf(s * 0.01f);
  }
  __syncthreads();
  for (int rr = wave * 2; rr < wave * 2 + 2; ++rr) {   // row squared norms
    float s = 0.f;
    for (int d = lane; d < DK_; d += 32) { float v = As[rr * DK_ + d]; s += v * v; }
    s = wred_sum(s);
    if (lane == 0) yq2_s[rr] = s;
  }
  __syncthreads();

  // ---- phase 2: GEMM1  S = yq(16x800) * xc^T(800x896), f32 WMMA ----
  // 56 N-tiles -> exactly 7 per wave, no guards -> EXEC stays all-ones.
  const float* xcT = xc + (size_t)t * NOP1 * DK_;
  {
    v8f acc[7];
    #pragma unroll
    for (int j = 0; j < 7; ++j) acc[j] = (v8f){};
    for (int kk = 0; kk < DK_; kk += 4) {
      v2f a = *(const v2f*)&As[llo * DK_ + kk + 2 * lhi];   // M=llo, K=kk+2*lhi
      #pragma unroll
      for (int j = 0; j < 7; ++j) {
        int o = (wave + 8 * j) * 16 + llo;
        v2f b = *(const v2f*)&xcT[(size_t)o * DK_ + kk + 2 * lhi];
        acc[j] = __builtin_amdgcn_wmma_f32_16x16x4_f32(
                     false, a, false, b, (short)0, acc[j], false, false);
      }
    }
    const float* xc2t = xc2 + t * NOP1;
    #pragma unroll
    for (int j = 0; j < 7; ++j) {
      int o = (wave + 8 * j) * 16 + llo;
      float x2 = xc2t[o];
      #pragma unroll
      for (int v = 0; v < 8; ++v) {                  // C layout: M = v + 8*lhi
        int r = v + 8 * lhi;
        float val = -INFINITY;
        if (o < NO_)
          val = -(yq2_s[r] + x2 - 2.f * acc[j][v]) / temp_s[r];
        Lg[r * NOP1 + o] = val;                      // cols >= 841 become -inf
      }
    }
  }
  __syncthreads();

  // ---- phase 3: K rounds of softmax-residual + GEMM2 (TDM-fed) + scatter ----
  const _Float16* xcpTt = xcpT + (size_t)t * ND2 * NOP2;
  const unsigned ldsBs = (unsigned)(unsigned long long)(void*)Bs; // LDS byte offset

  for (int k = 0; k < KK_; ++k) {
    // softmax round: each wave owns 2 rows
    for (int rr = wave * 2; rr < wave * 2 + 2; ++rr) {
      float* row = &Lg[rr * NOP1];
      float m = -INFINITY;
      for (int o = lane; o < NOP2; o += 32) m = fmaxf(m, row[o]);
      m = wred_max(m);
      float s = 0.f;
      for (int o = lane; o < NOP2; o += 32) s += __expf(row[o] - m);
      s = wred_sum(s);
      float lse = m + __logf(s);
      _Float16* wrow = &Wk[rr * NOP2];
      for (int o = lane; o < NOP2; o += 32) {
        float lp = row[o] - lse;
        wrow[o] = (_Float16)__expf(lp);
        float xm = fminf(lp, -1e-7f);                // log1mexp
        float upd = (xm < -0.6931472f) ? log1pf(-__expf(xm)) : __logf(-expm1f(xm));
        row[o] += upd;                               // -inf stays -inf
      }
    }

    v8f acc[7];
    #pragma unroll
    for (int j = 0; j < 7; ++j) acc[j] = (v8f){};

    for (int chunk = 0; chunk < 27; ++chunk) {
      __syncthreads();                               // prev reads done, Wk ready
      if (wave == 0) {
        // ---- Tensor Data Mover: 896x32 f16 tile, row stride 864 halves ----
        unsigned long long ga =
            (unsigned long long)(const void*)xcpTt + (unsigned long long)chunk * 64ull;
        u32x4 g0;
        g0[0] = 1u;                                  // count=1, user descriptor
        g0[1] = ldsBs;                               // lds_addr
        g0[2] = (unsigned)(ga & 0xffffffffu);        // global_addr[31:0]
        g0[3] = (unsigned)((ga >> 32) & 0x01ffffffu) // global_addr[56:32]
                | 0x80000000u;                       // type=2 ("image")
        i32x8 g1;
        g1[0] = 0x00010000;                          // data_size=1 -> 2 bytes
        g1[1] = (int)((unsigned)NOP2 << 16);         // tensor_dim0 = 864 (lo16)
        g1[2] = (int)((unsigned)ND2  << 16);         // dim0 hi=0 | tensor_dim1=896 (lo16)
        g1[3] = (int)(32u << 16);                    // dim1 hi=0 | tile_dim0 = 32
        g1[4] = ND2;                                 // tile_dim1 = 896, tile_dim2 = 0
        g1[5] = NOP2;                                // tensor_dim0_stride = 864 (lo32)
        g1[6] = 0;                                   // stride hi | dim1_stride lo
        g1[7] = 0;
        i32x4 gz = {0, 0, 0, 0};                     // 2D tensor: groups 2/3 unused
#if defined(__clang_major__) && (__clang_major__ >= 23)
        i32x8 gz8 = {0, 0, 0, 0, 0, 0, 0, 0};
        __builtin_amdgcn_tensor_load_to_lds(g0, g1, gz, gz, gz8, 0);
#else
        __builtin_amdgcn_tensor_load_to_lds(g0, g1, gz, gz, 0);
#endif
      }
      __builtin_amdgcn_s_wait_tensorcnt(0);          // TENSORcnt -> 0 (NOP for waves 1..7)
      __syncthreads();                               // Bs visible to all waves

      // A frag (W_k): M=llo, K halves [base..base+7] and [base+16..base+23]
      int obase = chunk * 32 + lhi * 8;
      union { uint4 u[2]; v16h h; } ua;
      ua.u[0] = *(const uint4*)&Wk[llo * NOP2 + obase];
      ua.u[1] = *(const uint4*)&Wk[llo * NOP2 + obase + 16];
      #pragma unroll
      for (int j = 0; j < 7; ++j) {                  // 56 d-tiles, uniform 7/wave
        int d = (wave + 8 * j) * 16 + llo;
        union { uint4 u[2]; v16h h; } ub;
        ub.u[0] = *(const uint4*)&Bs[d * 32 + lhi * 8];
        ub.u[1] = *(const uint4*)&Bs[d * 32 + lhi * 8 + 16];
        acc[j] = __builtin_amdgcn_wmma_f32_16x16x32_f16(
                     false, ua.h, false, ub.h, (short)0, acc[j], false, false);
      }
    }

    // scatter-add z[q,k,c,p] into vid[t][pix][k*8+c]
    #pragma unroll
    for (int j = 0; j < 7; ++j) {
      int n = (wave + 8 * j) * 16 + llo;
      if (n < DK_) {                                 // padded d-tiles contribute 0
        int cch = n / 100, p = n % 100;
        int dr = p / 10, dc = p % 10;
        #pragma unroll
        for (int v = 0; v < 8; ++v) {
          int q = qbase + v + 8 * lhi;
          if (q < NQ_) {
            int qh = q / NW_, qw = q % NW_;
            int pix = (qh + dr) * WP_ + (qw + dc);
            atomAddF(&vid[((size_t)t * (HP_ * WP_) + pix) * (KK_ * C_) + k * C_ + cch],
                     acc[j][v]);
          }
        }
      }
    }
    __syncthreads();                                 // Wk reuse next round
  }
}

// ---------------------------------------------------------------------------
// Finalize: out[t][ch][y][x]; coverage count is analytic (stride-1 patches).
// ---------------------------------------------------------------------------
__global__ __launch_bounds__(256) void n3_finalize_kernel(
    const float* __restrict__ x, const float* __restrict__ vid, float* __restrict__ out)
{
  int idx = blockIdx.x * 256 + threadIdx.x;
  if (idx >= T_ * 64 * H_ * W_) return;
  int xcol = idx % W_;
  int y    = (idx / W_) % H_;
  int ch   = (idx / (H_ * W_)) % 64;
  int t    = idx / (64 * H_ * W_);
  float r;
  if (ch < C_) {
    r = x[(((size_t)t * C_ + ch) * H_ + y) * W_ + xcol];
  } else {
    int cc = ch - C_;
    int k = cc >> 3, c = cc & 7;
    int py = y + 1, px = xcol + 1;
    int cny = min(py, 56) - max(py - 9, 0) + 1;
    int cnx = min(px, 56) - max(px - 9, 0) + 1;
    float cnt = (float)(cny * cnx);
    float zv = vid[((size_t)t * (HP_ * WP_) + py * WP_ + px) * (KK_ * C_) + k * C_ + c]
               / (cnt + 1e-10f);
    r = zv - x[(((size_t)t * C_ + c) * H_ + y) * W_ + xcol];
  }
  out[idx] = r;
}

// ---------------------------------------------------------------------------
extern "C" void kernel_launch(void* const* d_in, const int* in_sizes, int n_in,
                              void* d_out, int out_size, void* d_ws, size_t ws_size,
                              hipStream_t stream) {
  const float* x  = (const float*)d_in[0];
  const float* xe = (const float*)d_in[1];
  const float* ye = (const float*)d_in[2];
  const float* lt = (const float*)d_in[3];

  char* ws = (char*)d_ws;
  size_t off = 0;
  float* xc  = (float*)(ws + off);       off += (size_t)T_ * NOP1 * DK_ * 4;   // 5,734,400
  float* xc2 = (float*)(ws + off);       off += (size_t)T_ * NOP1 * 4;         //     7,168
  _Float16* xcpT = (_Float16*)(ws + off); off += (size_t)T_ * ND2 * NOP2 * 2;  // 3,096,576
  float* vid = (float*)(ws + off);
  size_t vidb = (size_t)T_ * HP_ * WP_ * KK_ * C_ * 4;                         // 1,951,488
  // total ws use ~10.8 MB

  hipMemsetAsync(vid, 0, vidb, stream);
  n3_prep_kernel<<<dim3(NOP1, T_), 256, 0, stream>>>(x, xe, xc, xc2, xcpT);
  n3_fused_kernel<<<dim3(NQT, T_), 256, 0, stream>>>(ye, lt, xc, xc2, xcpT, vid);
  n3_finalize_kernel<<<dim3((T_ * 64 * H_ * W_ + 255) / 256), 256, 0, stream>>>(
      x, vid, (float*)d_out);
}
